// SemanticSimilarityLoss_47442208751734
// MI455X (gfx1250) — compile-verified, matrix operand order checked by reading, not run on hardware
//
#include <hip/hip_runtime.h>

typedef __attribute__((ext_vector_type(16))) _Float16 v16h;
typedef __attribute__((ext_vector_type(8)))  _Float16 v8h;
typedef __attribute__((ext_vector_type(2)))  _Float16 v2h;
typedef __attribute__((ext_vector_type(8)))  float    v8f;

#define N_EMB  4096
#define D_EMB  512
#define C_CLS  1000
#define MARGIN 0.7f
#define EPS    1e-8f

// ---------------------------------------------------------------------------
// Kernel 1: L2-normalize each row of (N_EMB x D_EMB) f32 and convert to f16.
// One 256-thread block (8 waves) per row; each thread handles 2 elements.
// ---------------------------------------------------------------------------
__global__ __launch_bounds__(256)
void normalize_f16(const float* __restrict__ embs, _Float16* __restrict__ xh) {
    __shared__ float red[8];
    const int row = blockIdx.x;
    const int tid = threadIdx.x;
    const float* r = embs + (size_t)row * D_EMB;

    float2 v = *(const float2*)(r + tid * 2);
    float ss = v.x * v.x + v.y * v.y;
    #pragma unroll
    for (int o = 16; o > 0; o >>= 1) ss += __shfl_xor(ss, o, 32);

    const int wave = tid >> 5, lane = tid & 31;
    if (lane == 0) red[wave] = ss;
    __syncthreads();

    float tot = 0.0f;
    #pragma unroll
    for (int i = 0; i < 8; ++i) tot += red[i];

    const float scale = 1.0f / fmaxf(sqrtf(tot), EPS);
    v2h h;
    h.x = (_Float16)(v.x * scale);
    h.y = (_Float16)(v.y * scale);
    *(v2h*)(xh + (size_t)row * D_EMB + tid * 2) = h;
}

// ---------------------------------------------------------------------------
// WMMA fragment loader for the 16-bit 16x32 A layout (B = x^T loads the same
// way from row-major x). Lane L holds row base+(L&15); K chunks at
// kb + 8*(L>>4) and kb + 16 + 8*(L>>4), each an aligned 16-byte load.
// ---------------------------------------------------------------------------
__device__ __forceinline__ v16h load_frag(const _Float16* __restrict__ X,
                                          int baseRow, int kb, int lane) {
    const int r  = baseRow + (lane & 15);
    const int hi = lane >> 4;
    const _Float16* p = X + (size_t)r * D_EMB + kb + hi * 8;
    v8h c0 = *(const v8h*)(p);
    v8h c1 = *(const v8h*)(p + 16);
    return __builtin_shufflevector(c0, c1, 0,1,2,3,4,5,6,7,8,9,10,11,12,13,14,15);
}

// ---------------------------------------------------------------------------
// Kernel 2: sim = X * X^T via v_wmma_f32_16x16x32_f16, fused loss epilogue.
// Block = 256 threads = 8 waves; each wave owns a 32x32 region (2x2 tiles
// with A/B fragment reuse). Block covers 128x64; grid = 32 x 64.
// Per-block partial sums -> partials[] (deterministic, no float atomics).
// ---------------------------------------------------------------------------
__global__ __launch_bounds__(256)
void gemm_loss(const _Float16* __restrict__ X,
               const int* __restrict__ labels,
               const float* __restrict__ Aij,
               float* __restrict__ partials) {
    __shared__ float red[8];
    const int lane = threadIdx.x & 31;
    const int w    = threadIdx.x >> 5;
    const int wr   = w & 3;            // 4 waves along rows
    const int wc   = w >> 2;           // 2 waves along cols
    const int row0 = blockIdx.x * 128 + wr * 32;
    const int col0 = blockIdx.y * 64  + wc * 32;

    v8f c00 = {}, c01 = {}, c10 = {}, c11 = {};

    for (int kb = 0; kb < D_EMB; kb += 32) {
        v16h a0 = load_frag(X, row0,      kb, lane);
        v16h a1 = load_frag(X, row0 + 16, kb, lane);
        v16h b0 = load_frag(X, col0,      kb, lane);
        v16h b1 = load_frag(X, col0 + 16, kb, lane);
        c00 = __builtin_amdgcn_wmma_f32_16x16x32_f16(false, a0, false, b0, (short)0, c00, false, false);
        c01 = __builtin_amdgcn_wmma_f32_16x16x32_f16(false, a0, false, b1, (short)0, c01, false, false);
        c10 = __builtin_amdgcn_wmma_f32_16x16x32_f16(false, a1, false, b0, (short)0, c10, false, false);
        c11 = __builtin_amdgcn_wmma_f32_16x16x32_f16(false, a1, false, b1, (short)0, c11, false, false);
    }

    // Epilogue: C/D layout -> element (M = v + 8*(lane>>4), N = lane&15).
    const int hi   = lane >> 4;
    const int nIdx = lane & 15;
    const int lj0  = labels[col0 + nIdx];
    const int lj1  = labels[col0 + 16 + nIdx];

    float acc = 0.0f;
    #pragma unroll
    for (int ti = 0; ti < 2; ++ti) {
        #pragma unroll
        for (int v = 0; v < 8; ++v) {
            const int m  = row0 + ti * 16 + v + hi * 8;
            const int li = labels[m];
            const float* arow = Aij + (size_t)li * C_CLS;
            const float sim0 = ti ? c10[v] : c00[v];
            const float sim1 = ti ? c11[v] : c01[v];
            {
                const float de = 1.0f - sim0;
                const float dc = arow[lj0];
                if (dc < MARGIN && de < MARGIN) { const float t = de - dc; acc += t * t; }
            }
            {
                const float de = 1.0f - sim1;
                const float dc = arow[lj1];
                if (dc < MARGIN && de < MARGIN) { const float t = de - dc; acc += t * t; }
            }
        }
    }

    // wave reduce
    #pragma unroll
    for (int o = 16; o > 0; o >>= 1) acc += __shfl_xor(acc, o, 32);
    if (lane == 0) red[w] = acc;
    __syncthreads();
    if (threadIdx.x == 0) {
        float t = 0.0f;
        #pragma unroll
        for (int i = 0; i < 8; ++i) t += red[i];
        partials[blockIdx.y * gridDim.x + blockIdx.x] = t;
    }
}

// ---------------------------------------------------------------------------
// Kernel 3: deterministic fixed-order reduction of block partials, scale 1/N^2.
// ---------------------------------------------------------------------------
__global__ __launch_bounds__(256)
void final_reduce(const float* __restrict__ partials, int n, float* __restrict__ out) {
    __shared__ float s[256];
    float a = 0.0f;
    for (int i = threadIdx.x; i < n; i += 256) a += partials[i];
    s[threadIdx.x] = a;
    __syncthreads();
    #pragma unroll
    for (int st = 128; st > 0; st >>= 1) {
        if ((int)threadIdx.x < st) s[threadIdx.x] += s[threadIdx.x + st];
        __syncthreads();
    }
    if (threadIdx.x == 0)
        *out = s[0] * (1.0f / ((float)N_EMB * (float)N_EMB));
}

// ---------------------------------------------------------------------------
extern "C" void kernel_launch(void* const* d_in, const int* in_sizes, int n_in,
                              void* d_out, int out_size, void* d_ws, size_t ws_size,
                              hipStream_t stream) {
    const float* embs   = (const float*)d_in[0];   // (4096, 512) f32
    const int*   labels = (const int*)d_in[1];     // (4096,) int
    const float* Aij    = (const float*)d_in[2];   // (1000, 1000) f32
    float*       out    = (float*)d_out;           // scalar f32

    // workspace layout: [0, 4MB) f16 normalized embeddings; then block partials
    _Float16* xh = (_Float16*)d_ws;
    float* partials = (float*)((char*)d_ws + (size_t)N_EMB * D_EMB * sizeof(_Float16));

    normalize_f16<<<N_EMB, 256, 0, stream>>>(embs, xh);

    dim3 grid(32, 64); // 128-row x 64-col block tiles over 4096x4096
    gemm_loss<<<grid, 256, 0, stream>>>(xh, labels, Aij, partials);

    final_reduce<<<1, 256, 0, stream>>>(partials, 32 * 64, out);
}
